// AFTLayer_26577257627750
// MI455X (gfx1250) — matrix operand order repaired and checked by reading, not compiled, not run on hardware
//
#include <hip/hip_runtime.h>
#include <hip/hip_bf16.h>

#define NTOK 1728
#define CDIM 128
#define NB 4
#define NH 8
#define HD 16

typedef __attribute__((ext_vector_type(16))) __bf16        v16bf;
typedef __attribute__((ext_vector_type(8)))  float         v8f;
typedef __attribute__((ext_vector_type(4)))  unsigned int  u32x4;

union Frag16 { u32x4 q[2]; v16bf v; };

__device__ __forceinline__ unsigned short f2bf(float f) {
  unsigned int u = __float_as_uint(f);
  u += 0x7FFFu + ((u >> 16) & 1u);           // round-to-nearest-even
  return (unsigned short)(u >> 16);
}

// ---------------------------------------------------------------- convert
__global__ __launch_bounds__(256) void k_convert(
    const float* __restrict__ x,
    const float* __restrict__ Wq, const float* __restrict__ Wk,
    const float* __restrict__ Wv, const float* __restrict__ Wp,
    unsigned short* __restrict__ xb,
    unsigned short* __restrict__ wqb, unsigned short* __restrict__ wkb,
    unsigned short* __restrict__ wvb, unsigned short* __restrict__ wpb) {
  int t = blockIdx.x * blockDim.x + threadIdx.x;
  const int total = NB * NTOK * CDIM;
  for (int i = t; i < total; i += gridDim.x * blockDim.x) xb[i] = f2bf(x[i]);
  if (t < CDIM * CDIM) {
    wqb[t] = f2bf(Wq[t]); wkb[t] = f2bf(Wk[t]);
    wvb[t] = f2bf(Wv[t]); wpb[t] = f2bf(Wp[t]);
  }
}

// ---------------------------------------------------------------- q/k/v projection
// grid (108, 8), block 128 (4 waves). Wave w: row-tile mt = bx*4+w of B*N,
// col-tile = head h = by. Computes q,k,v 16x16 tiles sharing A fragments.
// Writes sigmoid(q) fp32 (B,H,N,16) and McombT[h][col][j] bf16 where
// col = b*32 + d -> ek*v, col = b*32+16+d -> ek   (kmax cancels in num/den).
__global__ __launch_bounds__(128) void k_proj(
    const unsigned short* __restrict__ xb,
    const unsigned short* __restrict__ wqb,
    const unsigned short* __restrict__ wkb,
    const unsigned short* __restrict__ wvb,
    float* __restrict__ qsig,
    unsigned short* __restrict__ mcomb) {
  const int lane = threadIdx.x & 31;
  const int wv_  = threadIdx.x >> 5;
  const int mt = blockIdx.x * 4 + wv_;
  const int h  = blockIdx.y;
  const int ln = lane & 15, kh = lane >> 4;

  v8f accq = {}, acck = {}, accv = {};
  const unsigned short* arow = xb  + (size_t)(mt * 16 + ln) * CDIM;
  const unsigned short* qrw  = wqb + (size_t)(h  * 16 + ln) * CDIM;
  const unsigned short* krw  = wkb + (size_t)(h  * 16 + ln) * CDIM;
  const unsigned short* vrw  = wvb + (size_t)(h  * 16 + ln) * CDIM;

  for (int kk = 0; kk < 4; ++kk) {
    const int k0 = kk * 32;
    Frag16 fa, fb;
    fa.q[0] = *(const u32x4*)(arow + k0 + kh * 8);
    fa.q[1] = *(const u32x4*)(arow + k0 + 16 + kh * 8);
    fb.q[0] = *(const u32x4*)(qrw + k0 + kh * 8);
    fb.q[1] = *(const u32x4*)(qrw + k0 + 16 + kh * 8);
    accq = __builtin_amdgcn_wmma_f32_16x16x32_bf16(false, fa.v, false, fb.v,
                                                   (short)0, accq, false, false);
    fb.q[0] = *(const u32x4*)(krw + k0 + kh * 8);
    fb.q[1] = *(const u32x4*)(krw + k0 + 16 + kh * 8);
    acck = __builtin_amdgcn_wmma_f32_16x16x32_bf16(false, fa.v, false, fb.v,
                                                   (short)0, acck, false, false);
    fb.q[0] = *(const u32x4*)(vrw + k0 + kh * 8);
    fb.q[1] = *(const u32x4*)(vrw + k0 + 16 + kh * 8);
    accv = __builtin_amdgcn_wmma_f32_16x16x32_bf16(false, fa.v, false, fb.v,
                                                   (short)0, accv, false, false);
  }

  const size_t hbase = (size_t)h * CDIM * NTOK;
  for (int r = 0; r < 8; ++r) {
    const int mg = mt * 16 + r + 8 * kh;
    const int b = mg / NTOK, i = mg - b * NTOK;
    const int d = ln;
    const float qs = 1.0f / (1.0f + __expf(-accq[r]));
    qsig[(((size_t)b * NH + h) * NTOK + i) * HD + d] = qs;
    const float ek  = __expf(acck[r]);
    const float ekv = ek * accv[r];
    mcomb[hbase + (size_t)(b * 32 + d)      * NTOK + i] = f2bf(ekv);
    mcomb[hbase + (size_t)(b * 32 + 16 + d) * NTOK + i] = f2bf(ek);
  }
}

// ---------------------------------------------------------------- fused bias-gather + GEMM
// grid (108 i-tiles, 8 heads), block 256 (8 waves, one 16x16 col-tile each).
// bmax cancels in num/den so eb = exp(bias) directly.
// 64-wide K chunks: 2 WMMAs + 4 ds_load_b128 + 4 global_load_b128 per wave
// per __syncthreads(), double-buffered LDS producer/consumer pipeline.
__global__ __launch_bounds__(256) void k_bias(
    const int* __restrict__ rel_index,
    const float* __restrict__ table,
    const unsigned short* __restrict__ mcomb,
    const float* __restrict__ qsig,
    unsigned short* __restrict__ attnb) {
  __shared__ unsigned short eb[2][16 * 64];   // bf16 eb tile, double buffered (4KB)
  __shared__ float res[16][128];              // num|den per batch (8KB)

  const int it = blockIdx.x, h = blockIdx.y;
  const int tid = threadIdx.x;
  const int lane = tid & 31, ct = tid >> 5;
  const int ln = lane & 15, kh = lane >> 4;

  // producer mapping: 16 rows x 16 j-pairs x 2 sub-chunks
  const int pi = tid >> 4;
  const int pj = (tid & 15) * 2;
  const int* __restrict__ idxrow = rel_index + (size_t)(it * 16 + pi) * NTOK;

  auto produce = [&](int c, int buf) {
    const int j0 = c * 64;
    __builtin_prefetch(idxrow + j0 + 128, 0, 1);   // global_prefetch_b8, two chunks ahead
    const int2 ia = *(const int2*)(idxrow + j0 + pj);
    const int2 ib = *(const int2*)(idxrow + j0 + 32 + pj);
    const float a0 = __expf(table[(size_t)ia.x * NH + h]);
    const float a1 = __expf(table[(size_t)ia.y * NH + h]);
    const float b0 = __expf(table[(size_t)ib.x * NH + h]);
    const float b1 = __expf(table[(size_t)ib.y * NH + h]);
    *(unsigned int*)&eb[buf][pi * 64 + pj] =
        (unsigned int)f2bf(a0) | ((unsigned int)f2bf(a1) << 16);
    *(unsigned int*)&eb[buf][pi * 64 + 32 + pj] =
        (unsigned int)f2bf(b0) | ((unsigned int)f2bf(b1) << 16);
  };

  const unsigned short* __restrict__ brow =
      mcomb + ((size_t)h * CDIM + ct * 16 + ln) * NTOK;

  v8f acc = {};
  produce(0, 0);
  for (int c = 0; c < 27; ++c) {
    __syncthreads();
    if (c + 1 < 27) produce(c + 1, (c + 1) & 1);
    const unsigned short* ebp = eb[c & 1];
    const int j0 = c * 64;
#pragma unroll
    for (int jj = 0; jj < 64; jj += 32) {
      Frag16 fa, fb;
      fa.q[0] = *(const u32x4*)(ebp + ln * 64 + jj + kh * 8);        // ds_load_b128
      fa.q[1] = *(const u32x4*)(ebp + ln * 64 + jj + 16 + kh * 8);
      fb.q[0] = *(const u32x4*)(brow + j0 + jj + kh * 8);            // global_load_b128
      fb.q[1] = *(const u32x4*)(brow + j0 + jj + 16 + kh * 8);
      acc = __builtin_amdgcn_wmma_f32_16x16x32_bf16(false, fa.v, false, fb.v,
                                                    (short)0, acc, false, false);
    }
  }

  for (int r = 0; r < 8; ++r)
    res[r + 8 * kh][ct * 16 + ln] = acc[r];
  __syncthreads();

  // out = sigmoid(q) * num/den  -> bf16 (B,N,C)
  for (int s = 0; s < 4; ++s) {
    const int e = tid + s * 256;              // 0..1023 = (b,i,d)
    const int b = e >> 8, rem = e & 255, i = rem >> 4, d = rem & 15;
    const float num = res[i][b * 32 + d];
    const float den = res[i][b * 32 + 16 + d];
    const float qs =
        qsig[(((size_t)b * NH + h) * NTOK + it * 16 + i) * HD + d];
    const float val = qs * (num / den);
    attnb[((size_t)(b * NTOK + it * 16 + i)) * CDIM + h * 16 + d] = f2bf(val);
  }
}

// ---------------------------------------------------------------- output projection
__global__ __launch_bounds__(128) void k_final(
    const unsigned short* __restrict__ attnb,
    const unsigned short* __restrict__ wpb,
    const float* __restrict__ bp,
    float* __restrict__ out) {
  const int lane = threadIdx.x & 31;
  const int wv_  = threadIdx.x >> 5;
  const int mt = blockIdx.x * 4 + wv_;
  const int nt = blockIdx.y;
  const int ln = lane & 15, kh = lane >> 4;

  v8f acc = {};
  const unsigned short* arow = attnb + (size_t)(mt * 16 + ln) * CDIM;
  const unsigned short* brow = wpb   + (size_t)(nt * 16 + ln) * CDIM;
  for (int kk = 0; kk < 4; ++kk) {
    const int k0 = kk * 32;
    Frag16 fa, fb;
    fa.q[0] = *(const u32x4*)(arow + k0 + kh * 8);
    fa.q[1] = *(const u32x4*)(arow + k0 + 16 + kh * 8);
    fb.q[0] = *(const u32x4*)(brow + k0 + kh * 8);
    fb.q[1] = *(const u32x4*)(brow + k0 + 16 + kh * 8);
    acc = __builtin_amdgcn_wmma_f32_16x16x32_bf16(false, fa.v, false, fb.v,
                                                  (short)0, acc, false, false);
  }
  const float bias = bp[nt * 16 + ln];
  for (int r = 0; r < 8; ++r) {
    const int mg = mt * 16 + r + 8 * kh;
    out[(size_t)mg * CDIM + nt * 16 + ln] = acc[r] + bias;
  }
}

// ---------------------------------------------------------------- launch
extern "C" void kernel_launch(void* const* d_in, const int* in_sizes, int n_in,
                              void* d_out, int out_size, void* d_ws, size_t ws_size,
                              hipStream_t stream) {
  const float* x     = (const float*)d_in[0];
  const float* Wq    = (const float*)d_in[1];
  const float* Wk    = (const float*)d_in[2];
  const float* Wv    = (const float*)d_in[3];
  const float* Wp    = (const float*)d_in[4];
  const float* bp    = (const float*)d_in[5];
  const float* table = (const float*)d_in[6];
  const int*   ridx  = (const int*)d_in[7];
  float* out = (float*)d_out;

  char* ws = (char*)d_ws;
  size_t o = 0;
  unsigned short* xb    = (unsigned short*)(ws + o); o += (size_t)NB * NTOK * CDIM * 2;  // 1.7MB
  unsigned short* wqb   = (unsigned short*)(ws + o); o += (size_t)CDIM * CDIM * 2;
  unsigned short* wkb   = (unsigned short*)(ws + o); o += (size_t)CDIM * CDIM * 2;
  unsigned short* wvb   = (unsigned short*)(ws + o); o += (size_t)CDIM * CDIM * 2;
  unsigned short* wpb   = (unsigned short*)(ws + o); o += (size_t)CDIM * CDIM * 2;
  float*          qsig  = (float*)         (ws + o); o += (size_t)NB * NTOK * CDIM * 4;  // 3.5MB
  unsigned short* mcomb = (unsigned short*)(ws + o); o += (size_t)NH * CDIM * NTOK * 2;  // 3.5MB
  unsigned short* attnb = (unsigned short*)(ws + o); o += (size_t)NB * NTOK * CDIM * 2;  // 1.7MB

  k_convert<<<dim3(432), 256, 0, stream>>>(x, Wq, Wk, Wv, Wp,
                                           xb, wqb, wkb, wvb, wpb);
  k_proj   <<<dim3(108, 8), 128, 0, stream>>>(xb, wqb, wkb, wvb, qsig, mcomb);
  k_bias   <<<dim3(108, 8), 256, 0, stream>>>(ridx, table, mcomb, qsig, attnb);
  k_final  <<<dim3(108, 8), 128, 0, stream>>>(attnb, wpb, bp, out);
}
